// EnhancedGNNModel_42709154791575
// MI455X (gfx1250) — matrix operand order repaired and checked by reading, not compiled, not run on hardware
//
#include <hip/hip_runtime.h>

typedef __attribute__((ext_vector_type(16))) __bf16          bf16x16;
typedef __attribute__((ext_vector_type(8)))  float           f32x8;
typedef __attribute__((ext_vector_type(4)))  float           f32x4;
typedef __attribute__((ext_vector_type(8)))  unsigned short  u16x8;

union BFrag {
    bf16x16        v;
    u16x8          h[2];
    unsigned short s[16];
};

__device__ inline unsigned short f2bf_rne(float f) {
    unsigned int u = __float_as_uint(f);
    unsigned int r = u + 0x7FFFu + ((u >> 16) & 1u);
    return (unsigned short)(r >> 16);
}

__device__ inline float bf2f(unsigned short h) {
    return __uint_as_float((unsigned int)h << 16);
}

// ---------------- utility kernels ----------------

__global__ void zero_f32v4_kernel(float* __restrict__ p, long long nVec4) {
    long long t = (long long)blockIdx.x * blockDim.x + threadIdx.x;
    if (t < nVec4) {
        f32x4 z; z[0] = 0.f; z[1] = 0.f; z[2] = 0.f; z[3] = 0.f;
        ((f32x4*)p)[t] = z;
    }
}

__global__ void zero_f32_kernel(float* __restrict__ p, long long n) {
    long long t = (long long)blockIdx.x * blockDim.x + threadIdx.x;
    if (t < n) p[t] = 0.0f;
}

__global__ void zero_u16_kernel(unsigned short* __restrict__ p, long long n) {
    long long t = (long long)blockIdx.x * blockDim.x + threadIdx.x;
    if (t < n) p[t] = 0;
}

__global__ void degree_kernel(const int* __restrict__ dst, float* __restrict__ deg, long long E) {
    long long t = (long long)blockIdx.x * blockDim.x + threadIdx.x;
    if (t < E)
        __hip_atomic_fetch_add(&deg[dst[t]], 1.0f, __ATOMIC_RELAXED, __HIP_MEMORY_SCOPE_AGENT);
}

__global__ void recip_kernel(float* __restrict__ deg, long long n) {
    long long t = (long long)blockIdx.x * blockDim.x + threadIdx.x;
    if (t < n) deg[t] = 1.0f / fmaxf(deg[t], 1.0f);
}

__global__ void f2bf_kernel(const float* __restrict__ in, unsigned short* __restrict__ out, long long n) {
    long long t = (long long)blockIdx.x * blockDim.x + threadIdx.x;
    if (t < n) out[t] = f2bf_rne(in[t]);
}

// concat [Wl | Wr] (each rows x 128, row-major) into bf16 [rowOff+rows, 256]
__global__ void wcat_kernel(const float* __restrict__ Wl, const float* __restrict__ Wr,
                            unsigned short* __restrict__ out, int rows, int rowOff) {
    int t = blockIdx.x * blockDim.x + threadIdx.x;
    if (t >= rows * 256) return;
    int r = t >> 8;
    int k = t & 255;
    float v = (k < 128) ? Wl[r * 128 + k] : Wr[r * 128 + (k - 128)];
    out[(size_t)(rowOff + r) * 256 + k] = f2bf_rne(v);
}

__global__ void head_bias_kernel(const float* __restrict__ ba, const float* __restrict__ bs,
                                 const float* __restrict__ be, float* __restrict__ out) {
    int j = threadIdx.x;
    if (j < 21)      out[j] = ba[j];
    else if (j < 23) out[j] = bs[j - 21];
    else if (j < 28) out[j] = be[j - 23];
    else if (j < 32) out[j] = 0.0f;
}

// ---------------- edge scatter (mean-agg numerator) ----------------
// 8 threads per edge, 16 channels each: gather bf16 features of src, atomic-add fp32 at dst.
__global__ void scatter_edges_kernel(const unsigned short* __restrict__ feat,
                                     const int* __restrict__ src,
                                     const int* __restrict__ dst,
                                     float* __restrict__ aggf, long long nWork) {
    long long t = (long long)blockIdx.x * blockDim.x + threadIdx.x;
    if (t >= nWork) return;
    long long e = t >> 3;
    int c = (int)(t & 7) * 16;
    int sN = src[e];
    int dN = dst[e];
    const unsigned short* p = feat + (size_t)sN * 128 + c;
    u16x8 lo = *(const u16x8*)p;
    u16x8 hi = *(const u16x8*)(p + 8);
    float* q = aggf + (size_t)dN * 128 + c;
#pragma unroll
    for (int i = 0; i < 8; ++i)
        __hip_atomic_fetch_add(q + i, bf2f(lo[i]), __ATOMIC_RELAXED, __HIP_MEMORY_SCOPE_AGENT);
#pragma unroll
    for (int i = 0; i < 8; ++i)
        __hip_atomic_fetch_add(q + 8 + i, bf2f(hi[i]), __ATOMIC_RELAXED, __HIP_MEMORY_SCOPE_AGENT);
}

// ---------------- fused SAGE GEMM: out = act([agg*inv | x] @ W^T + b) ----------------
// One wave per 16-row M-tile; NT n-tiles of 16 cols. K = 256 in 8 steps of 32 (bf16 WMMA).
// K-steps kept as rolled loops (#pragma unroll 1) so only one step's A/B fragments are
// live at once -> no VGPR spilling of B fragments.
template <int NT, bool RELU, bool OUTBF16>
__global__ __launch_bounds__(128, 1) void sage_gemm_kernel(
                                 const float* __restrict__ aggf,
                                 const float* __restrict__ inv_deg,
                                 const unsigned short* __restrict__ xr,   // bf16 [N,128]
                                 const unsigned short* __restrict__ W,    // bf16 [NT*16, 256]
                                 const float* __restrict__ bias,          // [NT*16]
                                 unsigned short* __restrict__ outb,       // bf16 [N, NT*16]
                                 float* __restrict__ outf,                // f32  [N, NT*16]
                                 int nTiles) {
    const int lane = threadIdx.x & 31;
    const int half = lane >> 4;
    const int lrow = lane & 15;
    const int wid    = (blockIdx.x * blockDim.x + threadIdx.x) >> 5;
    const int nWaves = (gridDim.x * blockDim.x) >> 5;

    for (int tile = wid; tile < nTiles; tile += nWaves) {
        const int   row = tile * 16 + lrow;
        const float s   = inv_deg[row];

        f32x8 acc[NT];
#pragma unroll
        for (int t = 0; t < NT; ++t)
#pragma unroll
            for (int i = 0; i < 8; ++i) acc[t][i] = 0.0f;

        // ---- K in [0,128): fp32 aggregate * inv_deg, converted to bf16 on the fly
        const float* arow = aggf + (size_t)row * 128 + half * 8;
        const unsigned short* wrow = W + (size_t)lrow * 256 + half * 8;
#pragma unroll 1
        for (int ks = 0; ks < 4; ++ks) {
            BFrag a;
            const float* p = arow + ks * 32;
#pragma unroll
            for (int i = 0; i < 8; ++i) a.s[i] = f2bf_rne(p[i] * s);
#pragma unroll
            for (int i = 0; i < 8; ++i) a.s[8 + i] = f2bf_rne(p[16 + i] * s);
#pragma unroll
            for (int t = 0; t < NT; ++t) {
                BFrag b;
                const unsigned short* wp = wrow + (size_t)t * 16 * 256 + ks * 32;
                b.h[0] = *(const u16x8*)wp;
                b.h[1] = *(const u16x8*)(wp + 16);
                acc[t] = __builtin_amdgcn_wmma_f32_16x16x32_bf16(
                    false, a.v, false, b.v, (short)0, acc[t], false, false);
            }
        }

        // ---- K in [128,256): bf16 activation rows
        const unsigned short* xrow = xr + (size_t)row * 128 + half * 8;
#pragma unroll 1
        for (int ks = 0; ks < 4; ++ks) {
            BFrag a;
            const unsigned short* p = xrow + ks * 32;
            a.h[0] = *(const u16x8*)p;
            a.h[1] = *(const u16x8*)(p + 16);
#pragma unroll
            for (int t = 0; t < NT; ++t) {
                BFrag b;
                const unsigned short* wp = wrow + (size_t)t * 16 * 256 + 128 + ks * 32;
                b.h[0] = *(const u16x8*)wp;
                b.h[1] = *(const u16x8*)(wp + 16);
                acc[t] = __builtin_amdgcn_wmma_f32_16x16x32_bf16(
                    false, a.v, false, b.v, (short)0, acc[t], false, false);
            }
        }

        // epilogue: C/D layout — VGPR i, lane l: M = i + 8*(l>=16), Ncol = l%16
#pragma unroll
        for (int t = 0; t < NT; ++t) {
            const int   col = t * 16 + lrow;
            const float bn  = bias[col];
#pragma unroll
            for (int i = 0; i < 8; ++i) {
                float v = acc[t][i] + bn;
                if (RELU) v = fmaxf(v, 0.0f);
                const size_t orow = (size_t)tile * 16 + i + half * 8;
                if (OUTBF16) outb[orow * (NT * 16) + col] = f2bf_rne(v);
                else         outf[orow * (NT * 16) + col] = v;
            }
        }
    }
}

// ---------------- split [N,32] head buffer into age/sex/eth regions ----------------
__global__ void split_heads_kernel(const float* __restrict__ headf, float* __restrict__ out, int N) {
    long long t = (long long)blockIdx.x * blockDim.x + threadIdx.x;
    if (t >= (long long)N * 28) return;
    int n = (int)(t / 28);
    int j = (int)(t % 28);
    float v = headf[(size_t)n * 32 + j];
    if (j < 21)      out[(size_t)n * 21 + j] = v;
    else if (j < 23) out[(size_t)N * 21 + (size_t)n * 2 + (j - 21)] = v;
    else             out[(size_t)N * 23 + (size_t)n * 5 + (j - 23)] = v;
}

// ---------------- host launch ----------------

static inline size_t align256(size_t x) { return (x + 255) & ~(size_t)255; }

extern "C" void kernel_launch(void* const* d_in, const int* in_sizes, int n_in,
                              void* d_out, int out_size, void* d_ws, size_t ws_size,
                              hipStream_t stream) {
    const float* x  = (const float*)d_in[0];
    const int*   ei = (const int*)d_in[1];
    const int       N = in_sizes[0] / 128;
    const long long E = (long long)in_sizes[1] / 2;
    const int* src = ei;
    const int* dst = ei + E;

    const float *Wl[6], *Wr[6], *bv[6];
    for (int i = 0; i < 6; ++i) {
        Wl[i] = (const float*)d_in[2 + 3 * i];
        Wr[i] = (const float*)d_in[3 + 3 * i];
        bv[i] = (const float*)d_in[4 + 3 * i];
    }

    // workspace carve
    char* w = (char*)d_ws;
    float*          aggf = (float*)w;           w += align256((size_t)N * 128 * 4);
    unsigned short* xb   = (unsigned short*)w;  w += align256((size_t)N * 128 * 2);
    unsigned short* hA   = (unsigned short*)w;  w += align256((size_t)N * 128 * 2);
    unsigned short* hB   = (unsigned short*)w;  w += align256((size_t)N * 128 * 2);
    float*          deg  = (float*)w;           w += align256((size_t)N * 4);
    unsigned short* wc0  = (unsigned short*)w;  w += align256((size_t)128 * 256 * 2);
    unsigned short* wc1  = (unsigned short*)w;  w += align256((size_t)128 * 256 * 2);
    unsigned short* wc2  = (unsigned short*)w;  w += align256((size_t)128 * 256 * 2);
    unsigned short* wh   = (unsigned short*)w;  w += align256((size_t)32 * 256 * 2);
    float*          hbias= (float*)w;           w += align256((size_t)32 * 4);
    float*          headf= (float*)w;           w += align256((size_t)N * 32 * 4);
    (void)ws_size; (void)n_in; (void)out_size;

    const int  B256    = 256;
    const long long nFeat = (long long)N * 128;
    const int  nTiles  = (N + 15) / 16;
    const int  gemmBlk = 128;                         // 4 waves / block
    const int  gemmGrd = (nTiles + 3) / 4;

    // degree + inverse
    zero_f32_kernel<<<(N + B256 - 1) / B256, B256, 0, stream>>>(deg, N);
    degree_kernel<<<(int)((E + B256 - 1) / B256), B256, 0, stream>>>(dst, deg, E);
    recip_kernel<<<(N + B256 - 1) / B256, B256, 0, stream>>>(deg, N);

    // bf16 conversions (x + weights)
    f2bf_kernel<<<(int)((nFeat + B256 - 1) / B256), B256, 0, stream>>>(x, xb, nFeat);
    wcat_kernel<<<(128 * 256 + B256 - 1) / B256, B256, 0, stream>>>(Wl[0], Wr[0], wc0, 128, 0);
    wcat_kernel<<<(128 * 256 + B256 - 1) / B256, B256, 0, stream>>>(Wl[1], Wr[1], wc1, 128, 0);
    wcat_kernel<<<(128 * 256 + B256 - 1) / B256, B256, 0, stream>>>(Wl[2], Wr[2], wc2, 128, 0);
    zero_u16_kernel<<<(32 * 256 + B256 - 1) / B256, B256, 0, stream>>>(wh, 32 * 256);
    wcat_kernel<<<(21 * 256 + B256 - 1) / B256, B256, 0, stream>>>(Wl[3], Wr[3], wh, 21, 0);
    wcat_kernel<<<(2 * 256 + B256 - 1) / B256, B256, 0, stream>>>(Wl[4], Wr[4], wh, 2, 21);
    wcat_kernel<<<(5 * 256 + B256 - 1) / B256, B256, 0, stream>>>(Wl[5], Wr[5], wh, 5, 23);
    head_bias_kernel<<<1, 32, 0, stream>>>(bv[3], bv[4], bv[5], hbias);

    const long long aggVec4 = nFeat / 4;
    const int zeroGrd = (int)((aggVec4 + B256 - 1) / B256);
    const long long scatWork = E * 8;
    const int scatGrd = (int)((scatWork + B256 - 1) / B256);

    // layer 1: xb -> hA
    zero_f32v4_kernel<<<zeroGrd, B256, 0, stream>>>(aggf, aggVec4);
    scatter_edges_kernel<<<scatGrd, B256, 0, stream>>>(xb, src, dst, aggf, scatWork);
    sage_gemm_kernel<8, true, true><<<gemmGrd, gemmBlk, 0, stream>>>(
        aggf, deg, xb, wc0, bv[0], hA, nullptr, nTiles);

    // layer 2: hA -> hB
    zero_f32v4_kernel<<<zeroGrd, B256, 0, stream>>>(aggf, aggVec4);
    scatter_edges_kernel<<<scatGrd, B256, 0, stream>>>(hA, src, dst, aggf, scatWork);
    sage_gemm_kernel<8, true, true><<<gemmGrd, gemmBlk, 0, stream>>>(
        aggf, deg, hA, wc1, bv[1], hB, nullptr, nTiles);

    // layer 3: hB -> hA
    zero_f32v4_kernel<<<zeroGrd, B256, 0, stream>>>(aggf, aggVec4);
    scatter_edges_kernel<<<scatGrd, B256, 0, stream>>>(hB, src, dst, aggf, scatWork);
    sage_gemm_kernel<8, true, true><<<gemmGrd, gemmBlk, 0, stream>>>(
        aggf, deg, hB, wc2, bv[2], hA, nullptr, nTiles);

    // heads (shared aggregation of hA): fused [N,256]x[256,32] GEMM, no relu, fp32 out
    zero_f32v4_kernel<<<zeroGrd, B256, 0, stream>>>(aggf, aggVec4);
    scatter_edges_kernel<<<scatGrd, B256, 0, stream>>>(hA, src, dst, aggf, scatWork);
    sage_gemm_kernel<2, false, false><<<gemmGrd, gemmBlk, 0, stream>>>(
        aggf, deg, hA, wh, hbias, nullptr, headf, nTiles);

    // scatter the 28 head columns into the three output regions
    const long long splitWork = (long long)N * 28;
    split_heads_kernel<<<(int)((splitWork + B256 - 1) / B256), B256, 0, stream>>>(
        headf, (float*)d_out, N);
}
